// PointCloudEncoder_56848187130596
// MI455X (gfx1250) — matrix operand order repaired and checked by reading, not compiled, not run on hardware
//
#include <hip/hip_runtime.h>
#include <cstdint>

typedef __attribute__((ext_vector_type(8)))  float          v8f;
typedef __attribute__((ext_vector_type(16))) __bf16         v16bf;
typedef __attribute__((ext_vector_type(8)))  unsigned short ushort8;
typedef __attribute__((ext_vector_type(16))) unsigned short ushort16;

#define KPAD 264      // bf16 LDS row stride (16B aligned, bank-conflict free)
#define CPAD 264      // f32 LDS row stride
#define OUTC 899
#define EPSF 1e-8f

__device__ __forceinline__ unsigned short f2bf(float f) {
    unsigned int u = __float_as_uint(f);
    u += 0x7FFFu + ((u >> 16) & 1u);          // round-to-nearest-even
    return (unsigned short)(u >> 16);
}

// ---------------- prep kernels ----------------

// weights -> bf16, transposed [N][Kp] (zero padded K)
__global__ void wconv_kernel(const float* __restrict__ w, unsigned short* __restrict__ dst,
                             int K, int C, int Kp) {
    int idx = blockIdx.x * blockDim.x + threadIdx.x;
    int tot = C * Kp;
    if (idx >= tot) return;
    int n = idx / Kp;
    int k = idx - n * Kp;
    dst[idx] = (k < K) ? f2bf(w[k * C + n]) : (unsigned short)0;
}

// P[b] = intrinsic[b] (3x3) @ extrinsic[b] (3x4)
__global__ void pmat_kernel(const float* __restrict__ extr, const float* __restrict__ intr,
                            float* __restrict__ P) {
    int t = threadIdx.x;
    if (t < 24) {
        int b = t / 12, r = (t % 12) / 4, c = t % 4;
        float s = 0.f;
        for (int k = 0; k < 3; ++k) s += intr[b * 9 + r * 3 + k] * extr[b * 12 + k * 4 + c];
        P[b * 12 + r * 4 + c] = s;
    }
}

// per (b, combined-channel) image mean and sqrt(var_unbiased + eps)
__global__ void imstats_kernel(const float* __restrict__ i0, const float* __restrict__ i1,
                               const float* __restrict__ i2,
                               float* __restrict__ imean, float* __restrict__ isd) {
    int bid = blockIdx.x;
    int b = bid / 448, cc = bid % 448;
    const float* img; int C, R, c0;
    if (cc < 64)       { img = i0; C = 64;  R = 64; c0 = cc; }
    else if (cc < 192) { img = i1; C = 128; R = 32; c0 = cc - 64; }
    else               { img = i2; C = 256; R = 16; c0 = cc - 192; }
    int HW = R * R;
    const float* p = img + (size_t)(b * C + c0) * HW;
    float s = 0.f, sq = 0.f;
    for (int i = threadIdx.x; i < HW; i += blockDim.x) { float v = p[i]; s += v; sq += v * v; }
    __shared__ float ss[256], s2[256];
    ss[threadIdx.x] = s; s2[threadIdx.x] = sq;
    __syncthreads();
    for (int o = 128; o > 0; o >>= 1) {
        if (threadIdx.x < o) { ss[threadIdx.x] += ss[threadIdx.x + o]; s2[threadIdx.x] += s2[threadIdx.x + o]; }
        __syncthreads();
    }
    if (threadIdx.x == 0) {
        float mean = ss[0] / (float)HW;
        float var  = (s2[0] - ss[0] * ss[0] / (float)HW) / (float)(HW - 1);
        imean[b * 448 + cc] = mean;
        isd  [b * 448 + cc] = sqrtf(var + EPSF);
    }
}

// ---------------- fragment loads ----------------

// A fragment: 16x32 bf16, lane M = lane&15; K runs {koff+half*8 .. +7, koff+16+half*8 .. +7}
__device__ __forceinline__ v16bf load_a(const unsigned short* aL, int lrow, int koff, int half) {
    ushort8 lo = *(const ushort8*)(aL + lrow * KPAD + koff + half * 8);
    ushort8 hi = *(const ushort8*)(aL + lrow * KPAD + koff + 16 + half * 8);
    ushort16 t;
#pragma unroll
    for (int i = 0; i < 8; ++i) { t[i] = lo[i]; t[i + 8] = hi[i]; }
    return __builtin_bit_cast(v16bf, t);
}

// B fragment: 32x16 bf16, lane N = lane&15; K runs {koff+half*8 .. +7, koff+16+half*8 .. +7} (wT[N][K])
__device__ __forceinline__ v16bf load_b(const unsigned short* wrow, int koff, int half) {
    ushort8 lo = *(const ushort8*)(wrow + koff + half * 8);
    ushort8 hi = *(const ushort8*)(wrow + koff + 16 + half * 8);
    ushort16 t;
#pragma unroll
    for (int i = 0; i < 8; ++i) { t[i] = lo[i]; t[i + 8] = hi[i]; }
    return __builtin_bit_cast(v16bf, t);
}

// ---------------- GEMM helpers ----------------

// D = relu(A @ W^T + bias); bf16 result -> outBf (next layer's A); optional f32 copy -> outF
__device__ __forceinline__ void gemm_relu(const unsigned short* aL, const unsigned short* wT,
                                          const float* bias, unsigned short* outBf, float* outF,
                                          int Cout, int K, int wave, int half, int lrow) {
    int ntiles = Cout >> 4;
    for (int nt = wave; nt < ntiles; nt += 4) {
        v8f acc = {0.f, 0.f, 0.f, 0.f, 0.f, 0.f, 0.f, 0.f};
        int ncol = nt * 16 + lrow;
        const unsigned short* wrow = wT + (size_t)ncol * K;
        for (int k = 0; k < K; k += 32) {
            v16bf a = load_a(aL, lrow, k, half);
            v16bf b = load_b(wrow, k, half);
            acc = __builtin_amdgcn_wmma_f32_16x16x32_bf16(false, a, false, b, (short)0, acc,
                                                          false, false);
        }
        float bv = bias[ncol];
#pragma unroll
        for (int r = 0; r < 8; ++r) {
            float v = acc[r] + bv;
            v = v > 0.f ? v : 0.f;
            outBf[(r + 8 * half) * KPAD + ncol] = f2bf(v);
            if (outF) outF[(r + 8 * half) * CPAD + ncol] = v;
        }
    }
}

__device__ __forceinline__ void gemm_fc_out(const unsigned short* aL, const unsigned short* wT,
                                            const float* bias, float* __restrict__ out,
                                            size_t m0, int choff,
                                            int Cout, int K, int wave, int half, int lrow) {
    int ntiles = Cout >> 4;
    for (int nt = wave; nt < ntiles; nt += 4) {
        v8f acc = {0.f, 0.f, 0.f, 0.f, 0.f, 0.f, 0.f, 0.f};
        int ncol = nt * 16 + lrow;
        const unsigned short* wrow = wT + (size_t)ncol * K;
        for (int k = 0; k < K; k += 32) {
            v16bf a = load_a(aL, lrow, k, half);
            v16bf b = load_b(wrow, k, half);
            acc = __builtin_amdgcn_wmma_f32_16x16x32_bf16(false, a, false, b, (short)0, acc,
                                                          false, false);
        }
        float bv = bias[ncol];
#pragma unroll
        for (int r = 0; r < 8; ++r)
            out[(m0 + (size_t)(r + 8 * half)) * OUTC + choff + ncol] = acc[r] + bv;
    }
}

// ---------------- main fused kernel ----------------

__global__ __launch_bounds__(128) void pce_main(
    const float* __restrict__ init_pc,
    const float* __restrict__ img0, const float* __restrict__ img1, const float* __restrict__ img2,
    const unsigned short* __restrict__ wT10, const float* __restrict__ b10,
    const unsigned short* __restrict__ wT20, const float* __restrict__ b20,
    const unsigned short* __restrict__ wT11, const float* __restrict__ b11,
    const unsigned short* __restrict__ wT21, const float* __restrict__ b21,
    const unsigned short* __restrict__ wT12, const float* __restrict__ b12,
    const unsigned short* __restrict__ wT22, const float* __restrict__ b22,
    const unsigned short* __restrict__ fT0, const float* __restrict__ fb0,
    const unsigned short* __restrict__ fT1, const float* __restrict__ fb1,
    const unsigned short* __restrict__ fT2, const float* __restrict__ fb2,
    const float* __restrict__ Pmat, const float* __restrict__ imean, const float* __restrict__ isd,
    float* __restrict__ out) {

    __shared__ __align__(16) unsigned short abfA[16 * KPAD];
    __shared__ __align__(16) unsigned short abfB[16 * KPAD];
    __shared__ __align__(16) unsigned short adbf[16 * KPAD];
    __shared__ __align__(16) float bufF[16 * CPAD];
    __shared__ float ps[16][8], psq[16][8];
    __shared__ float mu_s[16], rs_s[16];
    __shared__ float pcs[16][3];
    __shared__ float prj[3][16][6];

    const int tid = threadIdx.x;
    const int wave = tid >> 5, lane = tid & 31, half = lane >> 4, lrow = lane & 15;
    const int p8 = tid >> 3, sub8 = tid & 7;     // 8 threads per point
    const size_t m0 = (size_t)blockIdx.x * 16;
    const int b = (int)(m0 >> 16);               // 65536 points per batch

    // stage 0: load pc tile, zero-padded bf16 A (K=3 -> 32)
    for (int idx = tid; idx < 16 * 32; idx += 128) {
        int r = idx >> 5, c = idx & 31;
        float v = (c < 3) ? init_pc[(m0 + r) * 3 + c] : 0.f;
        abfA[r * KPAD + c] = f2bf(v);
        if (c < 3) pcs[r][c] = v;
    }
    __syncthreads();

    // per-point mean / 1/sqrt(var+eps) over channels of bufF (all 128 threads)
    auto stats = [&](int C) {
        float s = 0.f, sq = 0.f;
        for (int c = sub8; c < C; c += 8) {
            float v = bufF[p8 * CPAD + c];
            s += v; sq += v * v;
        }
        ps[p8][sub8] = s; psq[p8][sub8] = sq;
        __syncthreads();
        if (tid < 16) {
            float ts = 0.f, tq = 0.f;
#pragma unroll
            for (int i = 0; i < 8; ++i) { ts += ps[tid][i]; tq += psq[tid][i]; }
            float mu = ts / (float)C;
            float var = (tq - ts * ts / (float)C) / (float)(C - 1);
            mu_s[tid] = mu;
            rs_s[tid] = 1.0f / sqrtf(var + EPSF);
        }
    };
    auto buildAd = [&](int C, int cs, int lvoff) {
        for (int idx = tid; idx < 16 * C; idx += 128) {
            int r = idx >> cs, c = idx & (C - 1);
            float v = (bufF[r * CPAD + c] - mu_s[r]) * rs_s[r];
            v = (v + imean[b * 448 + lvoff + c]) * isd[b * 448 + lvoff + c];
            adbf[r * KPAD + c] = f2bf(v);
        }
    };

    // ---- block 0 ----
    gemm_relu(abfA, wT10, b10, abfB, nullptr, 64, 32, wave, half, lrow);   __syncthreads();
    gemm_relu(abfB, wT20, b20, abfA, bufF, 64, 64, wave, half, lrow);      __syncthreads(); // e0
    stats(64);                                                             __syncthreads();
    buildAd(64, 6, 0);                                                     __syncthreads();
    gemm_fc_out(adbf, fT0, fb0, out, m0, 0, 64, 64, wave, half, lrow);     __syncthreads();

    // ---- block 1 ----
    gemm_relu(abfA, wT11, b11, abfB, nullptr, 128, 64, wave, half, lrow);  __syncthreads();
    gemm_relu(abfB, wT21, b21, abfA, bufF, 128, 128, wave, half, lrow);    __syncthreads(); // e1
    stats(128);                                                            __syncthreads();
    buildAd(128, 7, 64);                                                   __syncthreads();
    gemm_fc_out(adbf, fT1, fb1, out, m0, 64, 128, 128, wave, half, lrow);  __syncthreads();

    // ---- block 2 ----
    gemm_relu(abfA, wT12, b12, abfB, nullptr, 256, 128, wave, half, lrow); __syncthreads();
    gemm_relu(abfB, wT22, b22, abfA, bufF, 256, 256, wave, half, lrow);    __syncthreads(); // e2
    stats(256);                                                            __syncthreads();
    buildAd(256, 8, 192);                                                  __syncthreads();
    gemm_fc_out(adbf, fT2, fb2, out, m0, 192, 256, 256, wave, half, lrow); __syncthreads();

    // ---- projection ----
    if (tid < 16) {
        float x = pcs[tid][0], y = pcs[tid][1], z = pcs[tid][2];
        const float* Pb = Pmat + b * 12;
        float X  = Pb[0] * x + Pb[1] * y + Pb[2]  * z + Pb[3];
        float Yc = Pb[4] * x + Pb[5] * y + Pb[6]  * z + Pb[7];
        float Z  = Pb[8] * x + Pb[9] * y + Pb[10] * z + Pb[11];
        float u = -X / Z, v = Yc / Z;
        const int Rs[3] = {64, 32, 16};
#pragma unroll
        for (int l = 0; l < 3; ++l) {
            float Rm1 = (float)(Rs[l] - 1);
            float ix = (u + 1.f) * 0.5f * Rm1; ix = fminf(fmaxf(ix, 0.f), Rm1);
            float iy = (v + 1.f) * 0.5f * Rm1; iy = fminf(fmaxf(iy, 0.f), Rm1);
            float x0 = floorf(ix), y0 = floorf(iy);
            prj[l][tid][0] = x0;
            prj[l][tid][1] = y0;
            prj[l][tid][2] = fminf(x0 + 1.f, Rm1);
            prj[l][tid][3] = fminf(y0 + 1.f, Rm1);
            prj[l][tid][4] = ix - x0;
            prj[l][tid][5] = iy - y0;
        }
    }
    __syncthreads();
    {
        const float* imgs[3] = {img0, img1, img2};
        const int Cs[3] = {64, 128, 256}, Rs[3] = {64, 32, 16}, off[3] = {448, 512, 640};
#pragma unroll
        for (int l = 0; l < 3; ++l) {
            int R = Rs[l], C = Cs[l];
            int x0 = (int)prj[l][p8][0], y0 = (int)prj[l][p8][1];
            int x1 = (int)prj[l][p8][2], y1 = (int)prj[l][p8][3];
            float wx = prj[l][p8][4], wy = prj[l][p8][5];
            float w00 = (1.f - wx) * (1.f - wy), w01 = wx * (1.f - wy);
            float w10 = (1.f - wx) * wy,         w11 = wx * wy;
            size_t orow = (m0 + p8) * OUTC + off[l];
            for (int c = sub8; c < C; c += 8) {
                const float* ch = imgs[l] + (size_t)(b * C + c) * R * R;
                float val = ch[y0 * R + x0] * w00 + ch[y0 * R + x1] * w01 +
                            ch[y1 * R + x0] * w10 + ch[y1 * R + x1] * w11;
                out[orow + c] = val;
            }
        }
    }
    if (tid < 48) {
        int p = tid / 3, k = tid - p * 3;
        out[(m0 + p) * OUTC + 896 + k] = pcs[p][k];
    }
}

// ---------------- host launch ----------------

extern "C" void kernel_launch(void* const* d_in, const int* in_sizes, int n_in,
                              void* d_out, int out_size, void* d_ws, size_t ws_size,
                              hipStream_t stream) {
    const float* init_pc = (const float*)d_in[0];
    const float* extr    = (const float*)d_in[1];
    const float* intr    = (const float*)d_in[2];
    const float* img0    = (const float*)d_in[3];
    const float* img1    = (const float*)d_in[4];
    const float* img2    = (const float*)d_in[5];
    const float* w1_0 = (const float*)d_in[6];  const float* b1_0 = (const float*)d_in[7];
    const float* w2_0 = (const float*)d_in[8];  const float* b2_0 = (const float*)d_in[9];
    const float* fcw0 = (const float*)d_in[10]; const float* fcb0 = (const float*)d_in[11];
    const float* w1_1 = (const float*)d_in[12]; const float* b1_1 = (const float*)d_in[13];
    const float* w2_1 = (const float*)d_in[14]; const float* b2_1 = (const float*)d_in[15];
    const float* fcw1 = (const float*)d_in[16]; const float* fcb1 = (const float*)d_in[17];
    const float* w1_2 = (const float*)d_in[18]; const float* b1_2 = (const float*)d_in[19];
    const float* w2_2 = (const float*)d_in[20]; const float* b2_2 = (const float*)d_in[21];
    const float* fcw2 = (const float*)d_in[22]; const float* fcb2 = (const float*)d_in[23];
    float* out = (float*)d_out;

    char* ws = (char*)d_ws;
    unsigned short* wT10 = (unsigned short*)(ws + 0);        // 64*32*2   = 4096
    unsigned short* wT20 = (unsigned short*)(ws + 4096);     // 64*64*2   = 8192
    unsigned short* wT11 = (unsigned short*)(ws + 12288);    // 128*64*2  = 16384
    unsigned short* wT21 = (unsigned short*)(ws + 28672);    // 128*128*2 = 32768
    unsigned short* wT12 = (unsigned short*)(ws + 61440);    // 256*128*2 = 65536
    unsigned short* wT22 = (unsigned short*)(ws + 126976);   // 256*256*2 = 131072
    unsigned short* fT0  = (unsigned short*)(ws + 258048);   // 8192
    unsigned short* fT1  = (unsigned short*)(ws + 266240);   // 32768
    unsigned short* fT2  = (unsigned short*)(ws + 299008);   // 131072 -> ends 430080
    float* Pm    = (float*)(ws + 430080);                    // 96 bytes (pad to 128)
    float* imean = (float*)(ws + 430208);                    // 2*448*4 = 3584
    float* isd   = (float*)(ws + 433792);                    // 3584 -> ends 437376

    auto cv = [&](const float* s, unsigned short* d, int K, int C, int Kp) {
        int tot = C * Kp;
        wconv_kernel<<<(tot + 255) / 256, 256, 0, stream>>>(s, d, K, C, Kp);
    };
    cv(w1_0, wT10, 3, 64, 32);
    cv(w2_0, wT20, 64, 64, 64);
    cv(w1_1, wT11, 64, 128, 64);
    cv(w2_1, wT21, 128, 128, 128);
    cv(w1_2, wT12, 128, 256, 128);
    cv(w2_2, wT22, 256, 256, 256);
    cv(fcw0, fT0, 64, 64, 64);
    cv(fcw1, fT1, 128, 128, 128);
    cv(fcw2, fT2, 256, 256, 256);

    pmat_kernel<<<1, 32, 0, stream>>>(extr, intr, Pm);
    imstats_kernel<<<896, 256, 0, stream>>>(img0, img1, img2, imean, isd);

    pce_main<<<8192, 128, 0, stream>>>(
        init_pc, img0, img1, img2,
        wT10, b1_0, wT20, b2_0,
        wT11, b1_1, wT21, b2_1,
        wT12, b1_2, wT22, b2_2,
        fT0, fcb0, fT1, fcb1, fT2, fcb2,
        Pm, imean, isd, out);
}
